// encoder_26465588478034
// MI455X (gfx1250) — matrix operand order repaired and checked by reading, not compiled, not run on hardware
//
#include <hip/hip_runtime.h>
#include <hip/hip_bf16.h>
#include <math.h>

// ---------------------------------------------------------------------------
// Bidirectional 2-layer LSTM encoder for MI455X (gfx1250, wave32, WMMA).
//  - All GEMMs via v_wmma_f32_16x16x32_f16 (fp16 in, fp32 accumulate).
//  - Recurrent scans: one 32-wave workgroup per direction; h/x in LDS,
//    per-wave LSTM cell state (c,h) kept in fp32 VGPR fragments.
//  - Layer-1 x tiles double-buffered via GLOBAL_LOAD_ASYNC_TO_LDS_B128
//    (ASYNCcnt) when the toolchain exposes the builtin; plain copy otherwise.
//  - global_prefetch for next-step gathered rows on the layer-0 critical path.
// ---------------------------------------------------------------------------

typedef _Float16 f16;
typedef __attribute__((ext_vector_type(16))) _Float16 v16h;
typedef __attribute__((ext_vector_type(8)))  _Float16 v8h;
typedef __attribute__((ext_vector_type(8)))  float    v8f;
typedef __attribute__((ext_vector_type(4)))  float    v4f;
typedef __attribute__((ext_vector_type(4)))  int      v4i;

union V16H { v16h v; v8h q[2]; f16 h[16]; };
union V8F  { v8f v; float f[8]; };
union V4F  { v4f v; float f[4]; };

#define S_ 512
#define B_ 32
#define E_ 1024
#define H_ 512

#if defined(__gfx1250__) && __has_builtin(__builtin_amdgcn_global_load_async_to_lds_b128) && \
    __has_builtin(__builtin_amdgcn_s_wait_asynccnt)
#define USE_ASYNC_LDS 1
typedef __attribute__((address_space(1))) v4i as1_v4i;
typedef __attribute__((address_space(3))) v4i as3_v4i;
#endif

__device__ __forceinline__ v8f wmma_f16(v16h a, v16h b, v8f c) {
  return __builtin_amdgcn_wmma_f32_16x16x32_f16(false, a, false, b, (short)0, c,
                                                false, false);
}

// A-operand (16x32 fp16) fragment from row-major LDS.
// ISA layout: lane m holds row m%16; lanes>=16 take the K+8/K+24 runs.
__device__ __forceinline__ v16h load_a_lds(const f16* base, int ld, int row0,
                                           int k0, int lane) {
  const f16* p = base + (size_t)(row0 + (lane & 15)) * ld + k0 + ((lane & 16) ? 8 : 0);
  V16H t;
  t.q[0] = *(const v8h*)p;         // K = kb+0 .. kb+7
  t.q[1] = *(const v8h*)(p + 16);  // K = kb+16 .. kb+23
  return t.v;
}

// B-operand (32x16 fp16) fragment from pre-packed tile buffer (512 halves/tile,
// stored in exact fragment order: tile[lane][half]).
__device__ __forceinline__ v16h load_b_tile(const f16* tiles, size_t tileIdx, int lane) {
  const f16* p = tiles + tileIdx * 512 + (size_t)lane * 16;
  V16H t;
  t.q[0] = *(const v8h*)p;
  t.q[1] = *(const v8h*)(p + 8);
  return t.v;
}

__device__ __forceinline__ float sigf(float x) { return 1.0f / (1.0f + __expf(-x)); }

// ---------------------------------------------------------------------------
// Weight packing: W (N x K, row-major fp32), interpret B[k][n] = W[n][k],
// tile (kt: K/32, nt: N/16), layout tileIdx = nt*(K/32)+kt, within tile
// lane%16 = n%16, lane half and element give K in table order (ISA B layout:
// N striped across lanes, K within a lane's halves).
// ---------------------------------------------------------------------------
__global__ __launch_bounds__(256) void pack_wb_kernel(const float* __restrict__ W,
                                                      f16* __restrict__ out,
                                                      int N, int K) {
  int tid = blockIdx.x * 256 + threadIdx.x;
  if (tid >= N * K) return;
  int tileIdx = tid >> 9;
  int within  = tid & 511;
  int lane = within >> 4, h = within & 15;
  int KT = K >> 5;
  int nt = tileIdx / KT, kt = tileIdx % KT;
  int kk = kt * 32 + (h & 7) + ((lane & 16) ? 8 : 0) + ((h & 8) ? 16 : 0);
  int n  = nt * 16 + (lane & 15);
  out[tid] = (f16)W[(size_t)n * K + kk];
}

// Combined [Wih ; Whh] pack: N=2048, K=1536 (first 1024 K from Wih, rest Whh).
__global__ __launch_bounds__(256) void pack_wb2_kernel(const float* __restrict__ Wih,
                                                       const float* __restrict__ Whh,
                                                       f16* __restrict__ out) {
  int tid = blockIdx.x * 256 + threadIdx.x;  // total 2048*1536
  int tileIdx = tid >> 9;
  int within  = tid & 511;
  int lane = within >> 4, h = within & 15;
  const int KT = 48;
  int nt = tileIdx / KT, kt = tileIdx % KT;
  int kk = kt * 32 + (h & 7) + ((lane & 16) ? 8 : 0) + ((h & 8) ? 16 : 0);
  int n  = nt * 16 + (lane & 15);
  float val = (kk < 1024) ? Wih[(size_t)n * 1024 + kk]
                          : Whh[(size_t)n * 512 + (kk - 1024)];
  out[tid] = (f16)val;
}

__global__ __launch_bounds__(256) void bias_kernel(const float* __restrict__ bih,
                                                   const float* __restrict__ bhh,
                                                   float* __restrict__ out) {
  int i = blockIdx.x * 256 + threadIdx.x;
  if (i < 8192) out[i] = bih[i] + bhh[i];
}

// ---------------------------------------------------------------------------
// Et tables: Et[e][s*B+b][:] = emb_e[tok(s,b)] @ W_et^T + b_et   (fp16 out)
// Parallel WMMA GEMM: M = S*B = 16384, K = 1024, N = 512.
// grid = (1024 M-tiles, 3 embeddings), 8 waves/block, 4 N-tiles/wave.
// ---------------------------------------------------------------------------
__global__ __launch_bounds__(256) void et_kernel(
    const int* __restrict__ seq, const float* __restrict__ emb1,
    const float* __restrict__ emb2, const float* __restrict__ emb3,
    const float* __restrict__ b_et, const f16* __restrict__ WBet,
    f16* __restrict__ Et) {
  const int mt = blockIdx.x;
  const int e  = blockIdx.y;
  const float* emb = (e == 0) ? emb1 : ((e == 1) ? emb2 : emb3);
  f16* outp = Et + (size_t)e * S_ * B_ * 512;
  const int w = threadIdx.x >> 5, lane = threadIdx.x & 31;
  const int lhalf = (lane & 16) ? 8 : 0;
  const int gr  = mt * 16 + (lane & 15);
  const int tok = seq[gr];
  const float* arow = emb + (size_t)tok * E_ + lhalf;

  V8F acc[4];
#pragma unroll
  for (int q = 0; q < 4; ++q)
#pragma unroll
    for (int v = 0; v < 8; ++v) acc[q].f[v] = 0.f;

  for (int kt = 0; kt < 32; ++kt) {
    const float* p = arow + kt * 32;
    V4F f0, f1, f2, f3;
    f0.v = *(const v4f*)p;        f1.v = *(const v4f*)(p + 4);
    f2.v = *(const v4f*)(p + 16); f3.v = *(const v4f*)(p + 20);
    V16H a;
#pragma unroll
    for (int i = 0; i < 4; ++i) {
      a.h[i]      = (f16)f0.f[i];
      a.h[4 + i]  = (f16)f1.f[i];
      a.h[8 + i]  = (f16)f2.f[i];
      a.h[12 + i] = (f16)f3.f[i];
    }
#pragma unroll
    for (int q = 0; q < 4; ++q) {
      v16h bf = load_b_tile(WBet, (size_t)(w * 4 + q) * 32 + kt, lane);
      acc[q].v = wmma_f16(a.v, bf, acc[q].v);
    }
  }
#pragma unroll
  for (int q = 0; q < 4; ++q) {
    int col = (w * 4 + q) * 16 + (lane & 15);
    float bb = b_et[col];
#pragma unroll
    for (int v = 0; v < 8; ++v) {
      int row = mt * 16 + v + lhalf;
      outp[(size_t)row * 512 + col] = (f16)(acc[q].f[v] + bb);
    }
  }
}

// ---------------------------------------------------------------------------
// Penalty: sum over masked tokens of relu(cos-0.3) terms. Symmetric under the
// sequence reversal, so compute once and double. Deterministic 2-stage reduce.
// ---------------------------------------------------------------------------
__global__ __launch_bounds__(256) void pen_partial_kernel(
    const int* __restrict__ seq, const int* __restrict__ lens,
    const float* __restrict__ e1, const float* __restrict__ e2,
    const float* __restrict__ e3, float* __restrict__ part) {
  const int s = blockIdx.x;
  const int w = threadIdx.x >> 5, lane = threadIdx.x & 31;
  __shared__ float sh[8];
  float acc = 0.f;
  for (int bi = 0; bi < 4; ++bi) {
    int b = w * 4 + bi;
    if (s < lens[b]) {
      int tok = seq[s * B_ + b];
      const float* p1 = e1 + (size_t)tok * E_;
      const float* p2 = e2 + (size_t)tok * E_;
      const float* p3 = e3 + (size_t)tok * E_;
      float d12 = 0, d13 = 0, d23 = 0, n1 = 0, n2 = 0, n3 = 0;
      for (int j = lane; j < E_; j += 32) {
        float a = p1[j], bb = p2[j], c = p3[j];
        d12 += a * bb; d13 += a * c; d23 += bb * c;
        n1 += a * a; n2 += bb * bb; n3 += c * c;
      }
#pragma unroll
      for (int off = 16; off > 0; off >>= 1) {
        d12 += __shfl_xor(d12, off, 32); d13 += __shfl_xor(d13, off, 32);
        d23 += __shfl_xor(d23, off, 32); n1 += __shfl_xor(n1, off, 32);
        n2 += __shfl_xor(n2, off, 32);   n3 += __shfl_xor(n3, off, 32);
      }
      if (lane == 0) {
        float s1 = sqrtf(n1), s2 = sqrtf(n2), s3 = sqrtf(n3);
        float c12 = d12 / fmaxf(s1 * s2, 1e-8f);
        float c13 = d13 / fmaxf(s1 * s3, 1e-8f);
        float c23 = d23 / fmaxf(s2 * s3, 1e-8f);
        acc += fmaxf(c12 - 0.3f, 0.f) + fmaxf(c13 - 0.3f, 0.f) + fmaxf(c23 - 0.3f, 0.f);
      }
    }
  }
  if (lane == 0) sh[w] = acc;
  __syncthreads();
  if (threadIdx.x == 0) {
    float t = 0.f;
    for (int i = 0; i < 8; ++i) t += sh[i];
    part[s] = t;
  }
}

__global__ __launch_bounds__(256) void pen_reduce_kernel(const float* __restrict__ part,
                                                         float* __restrict__ out) {
  __shared__ float sh[256];
  int tid = threadIdx.x;
  float a = 0.f;
  for (int i = tid; i < S_; i += 256) a += part[i];
  sh[tid] = a;
  __syncthreads();
  for (int s2 = 128; s2 > 0; s2 >>= 1) {
    if (tid < s2) sh[tid] += sh[tid + s2];
    __syncthreads();
  }
  if (tid == 0) out[0] = 2.0f * sh[0];  // forward + backward penalties are equal
}

// ---------------------------------------------------------------------------
// Layer-0 recurrent scan. One 1024-thread block per direction (blockIdx.x=d).
// Wave w owns hidden columns [w*16, w*16+16) and gate columns {j, j+H, j+2H,
// j+3H}; c,h kept as fp32 fragments in VGPRs. h (fp16) / x (fp16) in LDS.
// Next-step gathered rows (embeddings, Et) prefetched during current step.
// ---------------------------------------------------------------------------
__global__ __launch_bounds__(1024) void layer0_scan_kernel(
    const int* __restrict__ seq, const int* __restrict__ lens,
    const float* __restrict__ emb1, const float* __restrict__ emb2,
    const float* __restrict__ emb3, const f16* __restrict__ WBct,
    const f16* __restrict__ WB0, const float* __restrict__ biasLD,
    const f16* __restrict__ Et, const float* __restrict__ b_ct,
    const float* __restrict__ W_sc, const float* __restrict__ b_sc,
    const float* __restrict__ h0f, const float* __restrict__ c0f,
    const float* __restrict__ h0b, const float* __restrict__ c0b,
    f16* __restrict__ predF, f16* __restrict__ predB, f16* __restrict__ x1,
    float* __restrict__ out) {
  const int d    = blockIdx.x;
  const int tid  = threadIdx.x;
  const int w    = tid >> 5, lane = tid & 31;
  const int lhalf = (lane & 16) ? 8 : 0;
  const f16* WB   = WB0 + (size_t)d * (128ULL * 48ULL * 512ULL);
  const float* bias = biasLD + d * 2048;
  const float* h0 = d ? h0b : h0f;
  const float* c0 = d ? c0b : c0f;
  f16* pred = d ? predB : predF;

  extern __shared__ char smem[];
  f16* xs   = (f16*)smem;           // x  : 32 x 1032 fp16
  f16* hsh  = xs + 32 * 1032;       // h  : 32 x 520  fp16
  f16* cxs  = xs;                   // ctx: 32 x 520 fp16 (aliases x; dead before x written)
  float* wgt = (float*)(hsh + 32 * 520);  // softmax weights 32 x 4
  int* slen  = (int*)(wgt + 32 * 4);

  if (tid < 32) slen[tid] = lens[tid];
  const int colj = w * 16 + (lane & 15);
  for (int i = tid; i < 32 * 512; i += 1024) {
    int b = i >> 9, j = i & 511;
    hsh[b * 520 + j] = (f16)h0[j];
  }
  V8F hreg[2], creg[2];
#pragma unroll
  for (int mt = 0; mt < 2; ++mt)
#pragma unroll
    for (int v = 0; v < 8; ++v) {
      hreg[mt].f[v] = h0[colj];
      creg[mt].f[v] = c0[colj];
    }
  __syncthreads();

  for (int t = 0; t < S_; ++t) {
    // Phase 1: ctx = h @ W_ct^T + b_ct (wave w computes N-tile w)
    {
      V8F a0, a1;
#pragma unroll
      for (int v = 0; v < 8; ++v) { a0.f[v] = 0.f; a1.f[v] = 0.f; }
      for (int kt = 0; kt < 16; ++kt) {
        v16h af0 = load_a_lds(hsh, 520, 0, kt * 32, lane);
        v16h af1 = load_a_lds(hsh, 520, 16, kt * 32, lane);
        v16h bf  = load_b_tile(WBct, (size_t)w * 16 + kt, lane);
        a0.v = wmma_f16(af0, bf, a0.v);
        a1.v = wmma_f16(af1, bf, a1.v);
      }
      float bc = b_ct[colj];
#pragma unroll
      for (int v = 0; v < 8; ++v) {
        cxs[(v + lhalf) * 520 + colj]      = (f16)(a0.f[v] + bc);
        cxs[(16 + v + lhalf) * 520 + colj] = (f16)(a1.f[v] + bc);
      }
    }
    __syncthreads();
    // Phase 2: attention scores + softmax (wave w handles batch b = w)
    {
      const int b = w;
      const int len = slen[b];
      const int srow = (d == 0) ? t : ((t < len) ? (len - 1 - t) : t);
      float sc[3];
#pragma unroll
      for (int k3 = 0; k3 < 3; ++k3) {
        const f16* ep = Et + ((size_t)k3 * S_ * B_ + (size_t)srow * B_ + b) * 512 + lane * 16;
        const f16* cp = cxs + b * 520 + lane * 16;
        const float* wp = W_sc + lane * 16;
        float sum = 0.f;
#pragma unroll
        for (int jj = 0; jj < 16; ++jj)
          sum += tanhf((float)ep[jj] + (float)cp[jj]) * wp[jj];
#pragma unroll
        for (int off = 16; off > 0; off >>= 1) sum += __shfl_xor(sum, off, 32);
        sc[k3] = sum + b_sc[0];
      }
      // Prefetch next step's Et rows (gather-dependent; L2 can't predict them).
      if (t + 1 < S_) {
        int t2 = t + 1;
        int srow2 = (d == 0) ? t2 : ((t2 < len) ? (len - 1 - t2) : t2);
        const f16* nx = Et + ((size_t)srow2 * B_ + b) * 512 + lane * 16;
        __builtin_prefetch(nx, 0, 1);
        __builtin_prefetch(nx + (size_t)S_ * B_ * 512, 0, 1);
        __builtin_prefetch(nx + 2 * (size_t)S_ * B_ * 512, 0, 1);
      }
      if (lane == 0) {
        float mx = fmaxf(sc[0], fmaxf(sc[1], sc[2]));
        float e0 = __expf(sc[0] - mx), e1 = __expf(sc[1] - mx), e2 = __expf(sc[2] - mx);
        float inv = 1.f / (e0 + e1 + e2);
        wgt[b * 4 + 0] = e0 * inv;
        wgt[b * 4 + 1] = e1 * inv;
        wgt[b * 4 + 2] = e2 * inv;
      }
    }
    __syncthreads();
    // Phase 3: x = w1*e1 + w2*e2 + w3*e3 (fp16 -> LDS); write masked preds
    {
      const int b = tid >> 5;
      const int len = slen[b];
      const int srow = (d == 0) ? t : ((t < len) ? (len - 1 - t) : t);
      const int tok = seq[srow * B_ + b];
      const bool m = t < len;
      const float w1 = wgt[b * 4 + 0], w2 = wgt[b * 4 + 1], w3 = wgt[b * 4 + 2];
      const float* p1 = emb1 + (size_t)tok * E_ + lane * 32;
      const float* p2 = emb2 + (size_t)tok * E_ + lane * 32;
      const float* p3 = emb3 + (size_t)tok * E_ + lane * 32;
      f16* xr = xs + b * 1032 + lane * 32;
      f16* pr = pred + ((size_t)srow * B_ + b) * E_ + lane * 32;
#pragma unroll 8
      for (int jj = 0; jj < 32; ++jj) {
        float xv = w1 * p1[jj] + w2 * p2[jj] + w3 * p3[jj];
        xr[jj] = (f16)xv;
        pr[jj] = m ? (f16)xv : (f16)0.f;
      }
      // Prefetch next step's embedding rows while the gate GEMM runs.
      if (t + 1 < S_) {
        int t2 = t + 1;
        int srow2 = (d == 0) ? t2 : ((t2 < len) ? (len - 1 - t2) : t2);
        int tok2 = seq[srow2 * B_ + b];
        __builtin_prefetch(emb1 + (size_t)tok2 * E_ + lane * 32, 0, 1);
        __builtin_prefetch(emb2 + (size_t)tok2 * E_ + lane * 32, 0, 1);
        __builtin_prefetch(emb3 + (size_t)tok2 * E_ + lane * 32, 0, 1);
      }
    }
    __syncthreads();
    // Phase 4: gates = [x|h] @ [Wih;Whh]^T + bias  (wave w: gate cols g*512+w*16)
    V8F acc[4][2];
    {
#pragma unroll
      for (int g = 0; g < 4; ++g) {
        float bv = bias[g * 512 + colj];
#pragma unroll
        for (int mt = 0; mt < 2; ++mt)
#pragma unroll
          for (int v = 0; v < 8; ++v) acc[g][mt].f[v] = bv;
      }
      for (int kt = 0; kt < 48; ++kt) {
        v16h af0, af1;
        if (kt < 32) {
          af0 = load_a_lds(xs, 1032, 0, kt * 32, lane);
          af1 = load_a_lds(xs, 1032, 16, kt * 32, lane);
        } else {
          af0 = load_a_lds(hsh, 520, 0, (kt - 32) * 32, lane);
          af1 = load_a_lds(hsh, 520, 16, (kt - 32) * 32, lane);
        }
#pragma unroll
        for (int g = 0; g < 4; ++g) {
          v16h bf = load_b_tile(WB, (size_t)(g * 32 + w) * 48 + kt, lane);
          acc[g][0].v = wmma_f16(af0, bf, acc[g][0].v);
          acc[g][1].v = wmma_f16(af1, bf, acc[g][1].v);
        }
      }
    }
    __syncthreads();  // all waves done reading old h before we overwrite it
    // Phase 5: LSTM cell update; write h (LDS fp16), masked outs to x1
    {
#pragma unroll
      for (int mt = 0; mt < 2; ++mt) {
#pragma unroll
        for (int v = 0; v < 8; ++v) {
          int brow = mt * 16 + v + lhalf;
          int len = slen[brow];
          bool m = t < len;
          float iv = sigf(acc[0][mt].f[v]);
          float fv = sigf(acc[1][mt].f[v]);
          float gv = tanhf(acc[2][mt].f[v]);
          float ov = sigf(acc[3][mt].f[v]);
          float c2 = fv * creg[mt].f[v] + iv * gv;
          float h2 = ov * tanhf(c2);
          if (m) { creg[mt].f[v] = c2; hreg[mt].f[v] = h2; }
          hsh[brow * 520 + colj] = (f16)hreg[mt].f[v];
          int orow = (d == 0) ? t : ((t < len) ? (len - 1 - t) : t);
          x1[((size_t)orow * B_ + brow) * 1024 + d * 512 + colj] =
              m ? (f16)h2 : (f16)0.f;
          if (t == S_ - 1) {
            out[((size_t)d * B_ + brow) * H_ + colj] = hreg[mt].f[v];
            out[4 * B_ * H_ + ((size_t)d * B_ + brow) * H_ + colj] = creg[mt].f[v];
          }
        }
      }
    }
    __syncthreads();
  }
}

// ---------------------------------------------------------------------------
// Layer-1 recurrent scan. x tiles are double-buffered in LDS; step t+1's tile
// is fetched with GLOBAL_LOAD_ASYNC_TO_LDS_B128 (ASYNCcnt) while step t's
// gate GEMM runs, so the recurrence only pays s_wait_asynccnt at loop top.
// ---------------------------------------------------------------------------
__global__ __launch_bounds__(1024) void layer1_scan_kernel(
    const int* __restrict__ lens, const f16* __restrict__ x1,
    const f16* __restrict__ WB1, const float* __restrict__ biasLD,
    const float* __restrict__ h0f, const float* __restrict__ c0f,
    const float* __restrict__ h0b, const float* __restrict__ c0b,
    float* __restrict__ out) {
  const int d    = blockIdx.x;
  const int tid  = threadIdx.x;
  const int w    = tid >> 5, lane = tid & 31;
  const int lhalf = (lane & 16) ? 8 : 0;
  const f16* WB   = WB1 + (size_t)d * (128ULL * 48ULL * 512ULL);
  const float* bias = biasLD + (2 + d) * 2048;
  const float* h0 = d ? h0b : h0f;
  const float* c0 = d ? c0b : c0f;
  float* xout = out + 2 * 4 * B_ * H_;  // x output region

  extern __shared__ char smem[];
  f16* xb0 = (f16*)smem;        // x buffer 0: 32 x 1032
  f16* xb1 = xb0 + 32 * 1032;   // x buffer 1: 32 x 1032
  f16* hsh = xb1 + 32 * 1032;   // h: 32 x 520
  int* slen = (int*)(hsh + 32 * 520);

  if (tid < 32) slen[tid] = lens[tid];
  const int colj = w * 16 + (lane & 15);
  for (int i = tid; i < 32 * 512; i += 1024) {
    int b = i >> 9, j = i & 511;
    hsh[b * 520 + j] = (f16)h0[j];
  }
  V8F hreg[2], creg[2];
#pragma unroll
  for (int mt = 0; mt < 2; ++mt)
#pragma unroll
    for (int v = 0; v < 8; ++v) {
      hreg[mt].f[v] = h0[colj];
      creg[mt].f[v] = c0[colj];
    }
  __syncthreads();  // slen visible before tile-address computation

  const int myb = tid >> 5;  // batch row this thread stages
  // Stage the x tile for step tt into buf (64B per thread, 4x b128).
  auto stage_tile = [&](int tt, f16* buf) {
    int len = slen[myb];
    int srow = (d == 0) ? tt : ((tt < len) ? (len - 1 - tt) : tt);
    const f16* src = x1 + ((size_t)srow * B_ + myb) * 1024 + lane * 32;
    f16* dst = buf + myb * 1032 + lane * 32;
#ifdef USE_ASYNC_LDS
#pragma unroll
    for (int q = 0; q < 4; ++q)
      __builtin_amdgcn_global_load_async_to_lds_b128(
          (as1_v4i*)(src + q * 8), (as3_v4i*)(dst + q * 8), 0, 0);
#else
    const v8h* s8 = (const v8h*)src;
    v8h* d8 = (v8h*)dst;
#pragma unroll
    for (int q = 0; q < 4; ++q) d8[q] = s8[q];
#endif
  };

  stage_tile(0, xb0);  // preload tile for t=0

  for (int t = 0; t < S_; ++t) {
    f16* cur = (t & 1) ? xb1 : xb0;
    f16* nxt = (t & 1) ? xb0 : xb1;
#ifdef USE_ASYNC_LDS
    __builtin_amdgcn_s_wait_asynccnt(0);  // this wave's tile-t stores done
#endif
    __syncthreads();  // all waves: tile t resident in `cur`
    if (t + 1 < S_) stage_tile(t + 1, nxt);  // overlaps with GEMM below

    // Gate GEMM: gates = [x|h] @ [Wih;Whh]^T + bias
    V8F acc[4][2];
    {
#pragma unroll
      for (int g = 0; g < 4; ++g) {
        float bv = bias[g * 512 + colj];
#pragma unroll
        for (int mt = 0; mt < 2; ++mt)
#pragma unroll
          for (int v = 0; v < 8; ++v) acc[g][mt].f[v] = bv;
      }
      for (int kt = 0; kt < 48; ++kt) {
        v16h af0, af1;
        if (kt < 32) {
          af0 = load_a_lds(cur, 1032, 0, kt * 32, lane);
          af1 = load_a_lds(cur, 1032, 16, kt * 32, lane);
        } else {
          af0 = load_a_lds(hsh, 520, 0, (kt - 32) * 32, lane);
          af1 = load_a_lds(hsh, 520, 16, (kt - 32) * 32, lane);
        }
#pragma unroll
        for (int g = 0; g < 4; ++g) {
          v16h bf = load_b_tile(WB, (size_t)(g * 32 + w) * 48 + kt, lane);
          acc[g][0].v = wmma_f16(af0, bf, acc[g][0].v);
          acc[g][1].v = wmma_f16(af1, bf, acc[g][1].v);
        }
      }
    }
    __syncthreads();  // all waves done reading old h
    // Cell update; masked outs straight into d_out x region (fp32)
    {
#pragma unroll
      for (int mt = 0; mt < 2; ++mt) {
#pragma unroll
        for (int v = 0; v < 8; ++v) {
          int brow = mt * 16 + v + lhalf;
          int len = slen[brow];
          bool m = t < len;
          float iv = sigf(acc[0][mt].f[v]);
          float fv = sigf(acc[1][mt].f[v]);
          float gv = tanhf(acc[2][mt].f[v]);
          float ov = sigf(acc[3][mt].f[v]);
          float c2 = fv * creg[mt].f[v] + iv * gv;
          float h2 = ov * tanhf(c2);
          if (m) { creg[mt].f[v] = c2; hreg[mt].f[v] = h2; }
          hsh[brow * 520 + colj] = (f16)hreg[mt].f[v];
          int orow = (d == 0) ? t : ((t < len) ? (len - 1 - t) : t);
          xout[((size_t)orow * B_ + brow) * 1024 + d * 512 + colj] = m ? h2 : 0.f;
          if (t == S_ - 1) {
            out[((size_t)(2 + d) * B_ + brow) * H_ + colj] = hreg[mt].f[v];
            out[4 * B_ * H_ + ((size_t)(2 + d) * B_ + brow) * H_ + colj] = creg[mt].f[v];
          }
        }
      }
    }
    __syncthreads();
  }
}

// ---------------------------------------------------------------------------
// Prediction gating: gate = sigmoid([predF|predB] @ W_g^T + b_g),
// pred_emb = predF*gate + predB*(1-gate).  M=16384, K=2048, N=1024.
// ---------------------------------------------------------------------------
__global__ __launch_bounds__(256) void predgate_kernel(
    const f16* __restrict__ predF, const f16* __restrict__ predB,
    const f16* __restrict__ WBg, const float* __restrict__ b_g,
    float* __restrict__ outPE) {
  const int mt = blockIdx.x;
  const int w = threadIdx.x >> 5, lane = threadIdx.x & 31;
  const int lhalf = (lane & 16) ? 8 : 0;
  const int gr = mt * 16 + (lane & 15);
  const f16* pf = predF + (size_t)gr * 1024 + lhalf;
  const f16* pb = predB + (size_t)gr * 1024 + lhalf;

  V8F acc[8];
#pragma unroll
  for (int q = 0; q < 8; ++q)
#pragma unroll
    for (int v = 0; v < 8; ++v) acc[q].f[v] = 0.f;

  for (int kt = 0; kt < 64; ++kt) {
    const f16* src = (kt < 32) ? (pf + kt * 32) : (pb + (kt - 32) * 32);
    V16H a;
    a.q[0] = *(const v8h*)src;
    a.q[1] = *(const v8h*)(src + 16);
#pragma unroll
    for (int q = 0; q < 8; ++q) {
      v16h bf = load_b_tile(WBg, (size_t)(w * 8 + q) * 64 + kt, lane);
      acc[q].v = wmma_f16(a.v, bf, acc[q].v);
    }
  }
#pragma unroll
  for (int q = 0; q < 8; ++q) {
    int col = (w * 8 + q) * 16 + (lane & 15);
    float bg = b_g[col];
#pragma unroll
    for (int v = 0; v < 8; ++v) {
      int row = mt * 16 + v + lhalf;
      float g = sigf(acc[q].f[v] + bg);
      float a1 = (float)predF[(size_t)row * 1024 + col];
      float a2 = (float)predB[(size_t)row * 1024 + col];
      outPE[(size_t)row * 1024 + col] = a1 * g + a2 * (1.f - g);
    }
  }
}

// ---------------------------------------------------------------------------
extern "C" void kernel_launch(void* const* d_in, const int* in_sizes, int n_in,
                              void* d_out, int out_size, void* d_ws, size_t ws_size,
                              hipStream_t stream) {
  (void)in_sizes; (void)n_in; (void)out_size; (void)ws_size;
  const int*   seq  = (const int*)d_in[0];
  const int*   lens = (const int*)d_in[1];
  const float* emb1 = (const float*)d_in[2];
  const float* emb2 = (const float*)d_in[3];
  const float* emb3 = (const float*)d_in[4];
  const float* W_et = (const float*)d_in[5];
  const float* b_et = (const float*)d_in[6];
  const float* W_ct = (const float*)d_in[7];
  const float* b_ct = (const float*)d_in[8];
  const float* W_sc = (const float*)d_in[9];
  const float* b_sc = (const float*)d_in[10];
  const float* W_g  = (const float*)d_in[11];
  const float* b_g  = (const float*)d_in[12];
  const float* h0f  = (const float*)d_in[13];
  const float* c0f  = (const float*)d_in[14];
  const float* h0b  = (const float*)d_in[15];
  const float* c0b  = (const float*)d_in[16];
  const float* Wih  = (const float*)d_in[17];
  const float* Whh  = (const float*)d_in[18];
  const float* bih  = (const float*)d_in[19];
  const float* bhh  = (const float*)d_in[20];
  float* out = (float*)d_out;

  char* ws = (char*)d_ws;
  size_t off = 0;
  auto carve = [&](size_t bytes) -> char* {
    char* p = ws + off;
    off += (bytes + 255) & ~(size_t)255;
    return p;
  };
  f16* WBet = (f16*)carve((size_t)1024 * 512 * 2);
  f16* WBct = (f16*)carve((size_t)512 * 512 * 2);
  f16* WB0  = (f16*)carve((size_t)2 * 1536 * 2048 * 2);
  f16* WB1  = (f16*)carve((size_t)2 * 1536 * 2048 * 2);
  f16* WBg  = (f16*)carve((size_t)2048 * 1024 * 2);
  float* biasLD = (float*)carve((size_t)4 * 2048 * 4);
  f16* Et    = (f16*)carve((size_t)3 * S_ * B_ * 512 * 2);
  f16* predF = (f16*)carve((size_t)S_ * B_ * 1024 * 2);
  f16* predB = (f16*)carve((size_t)S_ * B_ * 1024 * 2);
  f16* x1    = (f16*)carve((size_t)S_ * B_ * 1024 * 2);
  float* pen = (float*)carve((size_t)S_ * 4);

  // Weight pack (fp32 -> fp16 WMMA-fragment tiles)
  pack_wb_kernel<<<dim3((512 * 1024) / 256), 256, 0, stream>>>(W_et, WBet, 512, 1024);
  pack_wb_kernel<<<dim3((512 * 512) / 256), 256, 0, stream>>>(W_ct, WBct, 512, 512);
  pack_wb_kernel<<<dim3((1024 * 2048) / 256), 256, 0, stream>>>(W_g, WBg, 1024, 2048);
  for (int l = 0; l < 2; ++l)
    for (int dd = 0; dd < 2; ++dd) {
      f16* dst = (l == 0 ? WB0 : WB1) + (size_t)dd * 1536 * 2048;
      pack_wb2_kernel<<<dim3((2048 * 1536) / 256), 256, 0, stream>>>(
          Wih + (size_t)(l * 2 + dd) * 2048 * 1024,
          Whh + (size_t)(l * 2 + dd) * 2048 * 512, dst);
    }
  bias_kernel<<<dim3(32), 256, 0, stream>>>(bih, bhh, biasLD);

  // Hoisted parallel WMMA GEMM + penalty
  et_kernel<<<dim3(1024, 3), 256, 0, stream>>>(seq, emb1, emb2, emb3, b_et, WBet, Et);
  pen_partial_kernel<<<dim3(S_), 256, 0, stream>>>(seq, lens, emb1, emb2, emb3, pen);
  pen_reduce_kernel<<<dim3(1), 256, 0, stream>>>(pen, out + 16908288);

  // Recurrent scans (one WGP-resident workgroup per direction)
  const size_t smem0 = (size_t)(32 * 1032 + 32 * 520) * 2 + 32 * 4 * 4 + 32 * 4 + 64;
  layer0_scan_kernel<<<dim3(2), 1024, smem0, stream>>>(
      seq, lens, emb1, emb2, emb3, WBct, WB0, biasLD, Et, b_ct, W_sc, b_sc,
      h0f, c0f, h0b, c0b, predF, predB, x1, out);
  const size_t smem1 = (size_t)(2 * 32 * 1032 + 32 * 520) * 2 + 32 * 4 + 64;
  layer1_scan_kernel<<<dim3(2), 1024, smem1, stream>>>(
      lens, x1, WB1, biasLD, h0f, c0f, h0b, c0b, out);

  // Final prediction gate (parallel WMMA GEMM)
  predgate_kernel<<<dim3(1024), 256, 0, stream>>>(predF, predB, WBg, b_g,
                                                  out + 16908289);
}
